// QuaMap1D_33200097198850
// MI455X (gfx1250) — compile-verified, matrix operand order checked by reading, not run on hardware
//
#include <hip/hip_runtime.h>

typedef __attribute__((ext_vector_type(16))) __bf16 v16bf;
typedef __attribute__((ext_vector_type(8)))  float  v8f;

#define N_TOK 256   // tokens per window (n)
#define CCH   128   // channels (c)
#define NS    4     // channel groups (s)
#define BW    2048  // b*w slices

__device__ __forceinline__ unsigned int f32_to_bf16_bits(float f) {
  unsigned int u = __float_as_uint(f);
  // round-to-nearest-even truncation to bf16
  return (u + 0x7FFFu + ((u >> 16) & 1u)) >> 16;
}

// ---------------------------------------------------------------------------
// Kernel 1: posmap[s][m][n] = softmax_n( d*c_s*sig_s - 0.5*sig_s*d^2 ), bf16.
// grid = 256 (one block per m), block = 256 (one thread per n).
// Output lives in workspace; 512 KB, L2-resident for the GEMM kernel.
// ---------------------------------------------------------------------------
__global__ void posmap_kernel(const float* __restrict__ centers,   // (4)
                              const float* __restrict__ spreads,   // (4)
                              unsigned short* __restrict__ pm)     // [4][256][256] bf16
{
  __shared__ float red[N_TOK];
  const int m = blockIdx.x;
  const int n = threadIdx.x;
  const float d  = (float)(n - m);
  const float d2 = d * d;
  for (int s = 0; s < NS; ++s) {
    const float sig = spreads[s];
    const float pp0 = centers[s] * sig;   // -0.5 * (-2*c*sig)
    const float pp1 = -0.5f * sig;
    const float score = d * pp0 + d2 * pp1;

    // max over n
    red[n] = score; __syncthreads();
    for (int off = 128; off > 0; off >>= 1) {
      if (n < off) red[n] = fmaxf(red[n], red[n + off]);
      __syncthreads();
    }
    const float mx = red[0]; __syncthreads();

    const float e = __expf(score - mx);
    red[n] = e; __syncthreads();
    for (int off = 128; off > 0; off >>= 1) {
      if (n < off) red[n] += red[n + off];
      __syncthreads();
    }
    const float inv = 1.0f / red[0]; __syncthreads();

    pm[((size_t)s * N_TOK + m) * N_TOK + n] =
        (unsigned short)f32_to_bf16_bits(e * inv);
  }
}

// ---------------------------------------------------------------------------
// Kernel 2: per bw-slice batched GEMM with V_WMMA_F32_16X16X32_BF16.
//   out[bw][m][c] = sum_n pm[s(c)][m][n] * x[bw][n][c]  + bias[m]
// grid = 2048 (one block per bw slice), block = 256 (8 waves).
//   - x slice (256x128 f32, 128 KB) staged to LDS as bf16 transposed [c][n]
//     (64 KB): B-fragment K-pairs are contiguous 32-bit LDS loads.
//   - bias (1 KB) staged to LDS: read back as 2x ds_load_b128 per tile.
//   - Wave w owns column group (s = w>>1, vt = w&1): 16 output channels,
//     sweeps 16 m-tiles in two halves of 8 accumulators (64 VGPRs).
//   - Direct scattered dword stores: the 8 waves jointly cover every channel
//     of the same rows, so partial-line writes merge in L2 before HBM.
// LDS 65 KB -> up to 4 blocks/WGP for latency hiding.
// ---------------------------------------------------------------------------
__global__ void __launch_bounds__(256, 1) qgemm_kernel(
    const float* __restrict__ x,            // [2048][256][128] f32
    const unsigned short* __restrict__ pm,  // [4][256][256] bf16
    const float* __restrict__ bias,         // [256]
    float* __restrict__ out)                // [2048][256][128] f32
{
  __shared__ unsigned short lds_x[CCH * N_TOK];   // [c][n] bf16, 64 KB
  __shared__ float          bias_s[N_TOK];        // 1 KB

  const int bw  = blockIdx.x;
  const int tid = threadIdx.x;
  const float* xs = x + (size_t)bw * (N_TOK * CCH);

  // ---- stage bias
  if (tid < N_TOK / 4) {
    *(float4*)&bias_s[tid * 4] = *(const float4*)(bias + tid * 4);
  }

  // ---- stage x: coalesced float4 global reads, cvt->bf16, transposed LDS
  {
    const int c4  = tid & 31;   // which float4 group along c
    const int np0 = tid >> 5;   // n-pair slice 0..7
    for (int it = 0; it < 16; ++it) {
      const int n0 = (np0 + it * 8) * 2;          // even token index
      const float4 a0 = *(const float4*)(xs + (size_t)(n0    ) * CCH + c4 * 4);
      const float4 a1 = *(const float4*)(xs + (size_t)(n0 + 1) * CCH + c4 * 4);
      const float f0[4] = {a0.x, a0.y, a0.z, a0.w};
      const float f1[4] = {a1.x, a1.y, a1.z, a1.w};
#pragma unroll
      for (int j = 0; j < 4; ++j) {
        const unsigned int packed = f32_to_bf16_bits(f0[j]) |
                                    (f32_to_bf16_bits(f1[j]) << 16);
        *(unsigned int*)&lds_x[(c4 * 4 + j) * N_TOK + n0] = packed;
      }
    }
  }
  __syncthreads();

  // ---- WMMA GEMM
  const int wave  = tid >> 5;        // 0..7
  const int lane  = tid & 31;
  const int s     = wave >> 1;       // channel group 0..3
  const int vt    = wave & 1;        // v-tile 0..1
  const int lrow  = lane & 15;       // A row / B col / D col within tile
  const int khalf = lane >> 4;       // which K half this lane holds
  const int c_out = (vt * 16 + lrow) * NS + s;            // output channel
  const unsigned short* pms = pm + (size_t)s * N_TOK * N_TOK;
  const unsigned short* ldsb = &lds_x[c_out * N_TOK];     // B column base
  float* outs = out + (size_t)bw * (N_TOK * CCH);

  for (int half = 0; half < 2; ++half) {         // m rows 0..127 / 128..255
    v8f acc[8] = {};
#pragma unroll
    for (int kk = 0; kk < 8; ++kk) {             // K steps of 32
      const int kb = kk * 32 + khalf * 8;
      // B fragment (x): k-pairs contiguous in LDS [c][n]
      union { unsigned int u[8]; v16bf v; } B;
#pragma unroll
      for (int p = 0; p < 8; ++p) {
        const int k0 = kb + ((p < 4) ? (2 * p) : (16 + 2 * (p - 4)));
        B.u[p] = *(const unsigned int*)&ldsb[k0];
      }
#pragma unroll
      for (int t = 0; t < 8; ++t) {              // 8 m-tiles share this B
        const int m = half * 128 + t * 16 + lrow;
        union { unsigned int u[8]; v16bf v; } A;
#pragma unroll
        for (int p = 0; p < 8; ++p) {
          const int k0 = kb + ((p < 4) ? (2 * p) : (16 + 2 * (p - 4)));
          A.u[p] = *(const unsigned int*)&pms[m * N_TOK + k0];
        }
        acc[t] = __builtin_amdgcn_wmma_f32_16x16x32_bf16(
            false, A.v, false, B.v, (short)0, acc[t], false, false);
      }
    }

    // ---- store with fused bias: per tile, the 8 bias values for this lane's
    //      rows are contiguous -> two b128 LDS loads instead of 8 vmem loads.
#pragma unroll
    for (int t = 0; t < 8; ++t) {
      const int mbase = half * 128 + t * 16 + khalf * 8;
      const float4 b0 = *(const float4*)&bias_s[mbase];
      const float4 b1 = *(const float4*)&bias_s[mbase + 4];
      const float bb[8] = {b0.x, b0.y, b0.z, b0.w, b1.x, b1.y, b1.z, b1.w};
#pragma unroll
      for (int e = 0; e < 8; ++e) {
        outs[(size_t)(mbase + e) * CCH + c_out] = acc[t][e] + bb[e];
      }
    }
  }
}

extern "C" void kernel_launch(void* const* d_in, const int* in_sizes, int n_in,
                              void* d_out, int out_size, void* d_ws, size_t ws_size,
                              hipStream_t stream) {
  const float* x       = (const float*)d_in[0];  // (16,128,256,128) f32
  const float* centers = (const float*)d_in[1];  // (4,1)
  const float* spreads = (const float*)d_in[2];  // (4,1)
  const float* bias    = (const float*)d_in[3];  // (1,256,1)
  float* out = (float*)d_out;
  unsigned short* pm = (unsigned short*)d_ws;    // 4*256*256 bf16 = 512 KB

  posmap_kernel<<<N_TOK, N_TOK, 0, stream>>>(centers, spreads, pm);
  qgemm_kernel<<<BW, 256, 0, stream>>>(x, pm, bias, out);
}
